// NodeDegreeExpansionBlock_7739531067655
// MI455X (gfx1250) — matrix operand order repaired and checked by reading, not compiled, not run on hardware
//
#include <hip/hip_runtime.h>
#include <hip/hip_bf16.h>
#include <math.h>

typedef __attribute__((ext_vector_type(16))) _Float16 v16h;
typedef __attribute__((ext_vector_type(8)))  float    v8f;
typedef __attribute__((ext_vector_type(2)))  float    v2f;

#define kN   4096
#define kG   64
#define kMH  128
#define kMU  64
#define kNSP 10
#define kEPS 1e-5f

#define INV_H  0.08838834764831845f   // 1/sqrt(128)
#define INV_U  0.125f                 // 1/sqrt(64)
#define INV_TP 0.0078125f             // 1/sqrt(2*128*64)
#define INV_SC 0.027950849718747374f  // 1/sqrt(128*10)
#define RS3    0.5773502691896258f    // 1/sqrt(3)

// ---------------------------------------------------------------------------
// CDNA5 async global->LDS staging (ASYNCcnt path), 16B per lane per call.
// ---------------------------------------------------------------------------
__device__ __forceinline__ void async_b128(void* lds_ptr, const void* gptr) {
  uint32_t lds = (uint32_t)(uintptr_t)lds_ptr;     // LDS aperture: low 32 bits
  asm volatile("global_load_async_to_lds_b128 %0, %1, off"
               :: "v"(lds), "v"(gptr) : "memory");
}
__device__ __forceinline__ void async_stage(const float* __restrict__ gsrc,
                                            float* lds_dst, int nfloats,
                                            int tid, int nthreads) {
  for (int off = tid * 4; off < nfloats; off += nthreads * 4)
    async_b128(lds_dst + off, gsrc + off);
}
__device__ __forceinline__ void async_wait_all() {
  asm volatile("s_wait_asynccnt 0x0" ::: "memory");
}

// ---------------------------------------------------------------------------
// f32 -> f16 weight conversion WITH fragment swizzle.
// Destination layout (fragment-major): for k-chunk kc (32 K), col-tile ct
// (16 cols), lane (0..31), j (0..15):
//   dst[((kc*NT + ct)*32 + lane)*16 + j] =
//        W[(kc*32 + 16*(lane>=16) + j) * ncols + ct*16 + (lane&15)]
// so each wave's B fragment is one contiguous, 32B-aligned v16h.
// ---------------------------------------------------------------------------
__global__ void cvt_swz_kernel(const float* __restrict__ src,
                               _Float16* __restrict__ dst,
                               int ncols, int total) {
  int e = blockIdx.x * 256 + threadIdx.x;
  if (e >= total) return;
  const int j    = e & 15;
  const int lane = (e >> 4) & 31;
  const int t    = e >> 9;            // kc*NT + ct
  const int NT   = ncols >> 4;
  const int ct   = t % NT;
  const int kc   = t / NT;
  const int k    = kc * 32 + ((lane & 16) ? 16 : 0) + j;
  const int col  = ct * 16 + (lane & 15);
  dst[e] = (_Float16)src[(size_t)k * ncols + col];
}

// ---------------------------------------------------------------------------
// Projection GEMM, full f32 precision via V_WMMA_F32_16X16X4_F32.
// C[M x ncols] = scale * A[M x K] @ B[K x ncols], A strided view.
// ---------------------------------------------------------------------------
__global__ __launch_bounds__(32)
void proj_kernel(const float* __restrict__ A, int lda, int astride,
                 const float* __restrict__ B, float* __restrict__ C,
                 int K, int ncols, float scale) {
  const int m0 = blockIdx.x * 16;
  const int n0 = blockIdx.y * 16;
  const int lane = threadIdx.x & 31;
  const int m  = lane & 15;
  const int nn = lane & 15;
  const int kb = (lane < 16) ? 0 : 2;

  v8f acc = {};
  for (int k = 0; k < K; k += 4) {
    v2f a, b;
    a[0] = A[(size_t)(m0 + m) * lda + (size_t)(k + kb + 0) * astride];
    a[1] = A[(size_t)(m0 + m) * lda + (size_t)(k + kb + 1) * astride];
    b[0] = B[(size_t)(k + kb + 0) * ncols + n0 + nn];
    b[1] = B[(size_t)(k + kb + 1) * ncols + n0 + nn];
    acc = __builtin_amdgcn_wmma_f32_16x16x4_f32(false, a, false, b,
                                                (short)0, acc, false, false);
  }
  const int mC = (lane < 16) ? 0 : 8;
#pragma unroll
  for (int r = 0; r < 8; ++r)
    C[(size_t)(m0 + mC + r) * ncols + n0 + nn] = acc[r] * scale;
}

// ---------------------------------------------------------------------------
// z0: K=8192 tensor-product GEMM (16 nodes x 256 cols per block).
// A fragments from register-resident f16 up-vectors x splat hp scalar;
// B fragments: one v16h load from pre-swizzled f16 weights.
// ---------------------------------------------------------------------------
__global__ __launch_bounds__(256)
void z0_kernel(const float* __restrict__ hp0, const float* __restrict__ up0,
               const float* __restrict__ hp1, const float* __restrict__ up1,
               const _Float16* __restrict__ W00s, const _Float16* __restrict__ W11s,
               float* __restrict__ z0) {
  __shared__ __align__(16) float sh_hp0[16 * 128];
  __shared__ __align__(16) float sh_up0[16 * 64];
  __shared__ __align__(16) float sh_hp1[3][16 * 128];
  __shared__ __align__(16) float sh_up1[3][16 * 64];

  const int n0 = blockIdx.x * 16;
  const int tid = threadIdx.x;

  // async stage the per-node-tile activations into LDS (ASYNCcnt path)
  async_stage(hp0 + (size_t)n0 * 128, sh_hp0, 16 * 128, tid, 256);
  async_stage(up0 + (size_t)n0 * 64,  sh_up0, 16 * 64,  tid, 256);
#pragma unroll
  for (int i = 0; i < 3; ++i) {
    async_stage(hp1 + ((size_t)i * kN + n0) * 128, sh_hp1[i], 16 * 128, tid, 256);
    async_stage(up1 + ((size_t)i * kN + n0) * 64,  sh_up1[i], 16 * 64,  tid, 256);
  }
  async_wait_all();
  __syncthreads();

  const int wave = tid >> 5, lane = tid & 31;
  const int m   = lane & 15;
  const int kbA = (lane < 16) ? 0 : 16;
  const int mC  = (lane < 16) ? 0 : 8;
  const int nn  = lane & 15;

  // hoist this lane's up-vector slices into f16 register fragments
  v16h u0f[2], u1f[3][2];
#pragma unroll
  for (int half = 0; half < 2; ++half) {
#pragma unroll
    for (int j = 0; j < 16; ++j) {
      const int vv = half * 32 + kbA + j;
      u0f[half][j] = (_Float16)sh_up0[m * 64 + vv];
      u1f[0][half][j] = (_Float16)sh_up1[0][m * 64 + vv];
      u1f[1][half][j] = (_Float16)sh_up1[1][m * 64 + vv];
      u1f[2][half][j] = (_Float16)sh_up1[2][m * 64 + vv];
    }
  }

  for (int ct = wave; ct < 16; ct += 8) {
    const v16h* pB00 = (const v16h*)W00s + (size_t)ct * 32 + lane;
    const v16h* pB11 = (const v16h*)W11s + (size_t)ct * 32 + lane;
    v8f acc = {};
    for (int u = 0; u < 128; ++u) {
      const _Float16 a0 = (_Float16)sh_hp0[m * 128 + u];
      const _Float16 c0 = (_Float16)(sh_hp1[0][m * 128 + u] * RS3);
      const _Float16 c1 = (_Float16)(sh_hp1[1][m * 128 + u] * RS3);
      const _Float16 c2 = (_Float16)(sh_hp1[2][m * 128 + u] * RS3);
#pragma unroll
      for (int half = 0; half < 2; ++half) {
        const int kc = u * 2 + half;
        v16h A00 = u0f[half] * a0;
        v16h A11 = u1f[0][half] * c0 + u1f[1][half] * c1 + u1f[2][half] * c2;
        const v16h B00 = pB00[(size_t)kc * 512];   // 16 col-tiles * 32 lanes
        const v16h B11 = pB11[(size_t)kc * 512];
        acc = __builtin_amdgcn_wmma_f32_16x16x32_f16(false, A00, false, B00,
                                                     (short)0, acc, false, false);
        acc = __builtin_amdgcn_wmma_f32_16x16x32_f16(false, A11, false, B11,
                                                     (short)0, acc, false, false);
      }
    }
    const int cb = ct * 16;
#pragma unroll
    for (int r = 0; r < 8; ++r)
      z0[(size_t)(n0 + mC + r) * 256 + cb + nn] = acc[r] * INV_TP;
  }
}

// ---------------------------------------------------------------------------
// z1 (per i): same structure, 128 output cols, weights W01/W10 (swizzled).
// grid = (N/16, 3); 8 waves = 8 col tiles.
// ---------------------------------------------------------------------------
__global__ __launch_bounds__(256)
void z1_kernel(const float* __restrict__ hp0, const float* __restrict__ up0,
               const float* __restrict__ hp1, const float* __restrict__ up1,
               const _Float16* __restrict__ W01s, const _Float16* __restrict__ W10s,
               float* __restrict__ z1) {
  __shared__ __align__(16) float sh_hp0[16 * 128];
  __shared__ __align__(16) float sh_up0[16 * 64];
  __shared__ __align__(16) float sh_hp1[16 * 128];
  __shared__ __align__(16) float sh_up1[16 * 64];

  const int n0 = blockIdx.x * 16;
  const int i  = blockIdx.y;
  const int tid = threadIdx.x;

  async_stage(hp0 + (size_t)n0 * 128, sh_hp0, 16 * 128, tid, 256);
  async_stage(up0 + (size_t)n0 * 64,  sh_up0, 16 * 64,  tid, 256);
  async_stage(hp1 + ((size_t)i * kN + n0) * 128, sh_hp1, 16 * 128, tid, 256);
  async_stage(up1 + ((size_t)i * kN + n0) * 64,  sh_up1, 16 * 64,  tid, 256);
  async_wait_all();
  __syncthreads();

  const int wave = tid >> 5, lane = tid & 31;
  const int m   = lane & 15;
  const int kbA = (lane < 16) ? 0 : 16;
  const int mC  = (lane < 16) ? 0 : 8;
  const int nn  = lane & 15;

  v16h u0f[2], u1f[2];
#pragma unroll
  for (int half = 0; half < 2; ++half) {
#pragma unroll
    for (int j = 0; j < 16; ++j) {
      const int vv = half * 32 + kbA + j;
      u0f[half][j] = (_Float16)sh_up0[m * 64 + vv];
      u1f[half][j] = (_Float16)sh_up1[m * 64 + vv];
    }
  }

  const int ct = wave;                 // 8 waves x 16 = 128 cols
  const v16h* pB01 = (const v16h*)W01s + (size_t)ct * 32 + lane;
  const v16h* pB10 = (const v16h*)W10s + (size_t)ct * 32 + lane;

  v8f acc = {};
  for (int u = 0; u < 128; ++u) {
    const _Float16 a0 = (_Float16)sh_hp0[m * 128 + u];   // pairs with up1_i
    const _Float16 a1 = (_Float16)sh_hp1[m * 128 + u];   // pairs with up0
#pragma unroll
    for (int half = 0; half < 2; ++half) {
      const int kc = u * 2 + half;
      v16h A0 = u1f[half] * a0;
      v16h A1 = u0f[half] * a1;
      const v16h B0 = pB01[(size_t)kc * 256];   // 8 col-tiles * 32 lanes
      const v16h B1 = pB10[(size_t)kc * 256];
      acc = __builtin_amdgcn_wmma_f32_16x16x32_f16(false, A0, false, B0,
                                                   (short)0, acc, false, false);
      acc = __builtin_amdgcn_wmma_f32_16x16x32_f16(false, A1, false, B1,
                                                   (short)0, acc, false, false);
    }
  }
  const int cb = ct * 16;
#pragma unroll
  for (int r = 0; r < 8; ++r)
    z1[((size_t)i * kN + n0 + mC + r) * 128 + cb + nn] = acc[r] * INV_TP;
}

// ---------------------------------------------------------------------------
// post: s=silu(z0[:,:128]), g=sigmoid(z0[:,128:]), v=z1*g;
// y0 = s@W_p0*INV_H + b_p0 + h0@Wsc0[:,sp,:]*INV_SC
// y1_i = v_i@W_p1*INV_H + h1_i@Wsc1[:,sp,:]*INV_SC
// ---------------------------------------------------------------------------
__global__ __launch_bounds__(128)
void post_kernel(const float* __restrict__ nfh, const float* __restrict__ onehot,
                 const float* __restrict__ z0, const float* __restrict__ z1,
                 const float* __restrict__ W_p0, const float* __restrict__ b_p0,
                 const float* __restrict__ W_p1, const float* __restrict__ Wsc0,
                 const float* __restrict__ Wsc1,
                 float* __restrict__ y0, float* __restrict__ y1) {
  const int n = blockIdx.x, w = threadIdx.x;
  __shared__ float s_l[128], g_l[128], v_l[3][128], h0_l[128], h1_l[3][128];
  __shared__ int sp_s;

  float zs = z0[(size_t)n * 256 + w];
  s_l[w] = zs / (1.f + __expf(-zs));
  float zg = z0[(size_t)n * 256 + 128 + w];
  float g = 1.f / (1.f + __expf(-zg));
  g_l[w] = g;
  h0_l[w] = nfh[(size_t)n * 512 + w];
#pragma unroll
  for (int i = 0; i < 3; ++i) {
    v_l[i][w] = z1[((size_t)i * kN + n) * 128 + w] * g;
    h1_l[i][w] = nfh[(size_t)n * 512 + 128 + w * 3 + i];
  }
  if (w == 0) {
    int sp = 0;
    for (int t = 0; t < kNSP; ++t)
      if (onehot[(size_t)n * kNSP + t] > 0.5f) sp = t;
    sp_s = sp;
  }
  __syncthreads();
  const int sp = sp_s;

  float acc0 = 0.f, accs = 0.f;
  float acc1[3] = {0.f, 0.f, 0.f}, accs1[3] = {0.f, 0.f, 0.f};
  for (int u = 0; u < 128; ++u) {
    const float wp0 = W_p0[(size_t)u * 128 + w];
    const float wc0 = Wsc0[((size_t)u * kNSP + sp) * 128 + w];
    const float wp1 = W_p1[(size_t)u * 128 + w];
    const float wc1 = Wsc1[((size_t)u * kNSP + sp) * 128 + w];
    acc0 = fmaf(s_l[u], wp0, acc0);
    accs = fmaf(h0_l[u], wc0, accs);
#pragma unroll
    for (int i = 0; i < 3; ++i) {
      acc1[i]  = fmaf(v_l[i][u], wp1, acc1[i]);
      accs1[i] = fmaf(h1_l[i][u], wc1, accs1[i]);
    }
  }
  y0[(size_t)n * 128 + w] = acc0 * INV_H + b_p0[w] + accs * INV_SC;
#pragma unroll
  for (int i = 0; i < 3; ++i)
    y1[((size_t)i * kN + n) * 128 + w] = acc1[i] * INV_H + accs1[i] * INV_SC;
}

// ---------------------------------------------------------------------------
// Group statistics. stats: [0..63]=cnt, [64..127]=sum0->mean0,
// [128..191]=varsum0->rstd0, [192..255]=varsum1->rstd1
// ---------------------------------------------------------------------------
__global__ void stats_zero_kernel(float* __restrict__ stats) {
  if (threadIdx.x < 4 * kG) stats[threadIdx.x] = 0.f;
}

__global__ __launch_bounds__(128)
void stats1_kernel(const float* __restrict__ y0, const int* __restrict__ batch,
                   float* __restrict__ stats) {
  const int n = blockIdx.x, w = threadIdx.x;
  __shared__ float red[128];
  red[w] = y0[(size_t)n * 128 + w];
  __syncthreads();
  for (int s = 64; s > 0; s >>= 1) {
    if (w < s) red[w] += red[w + s];
    __syncthreads();
  }
  if (w == 0) {
    const int g = batch[n];
    atomicAdd(&stats[g], 1.0f);
    atomicAdd(&stats[kG + g], red[0]);
  }
}

__global__ void stats_mid_kernel(float* __restrict__ stats) {
  const int g = threadIdx.x;
  if (g >= kG) return;
  float c = fmaxf(stats[g], 1.f);
  stats[g] = c;
  stats[kG + g] = stats[kG + g] / (c * 128.f);
}

__global__ __launch_bounds__(128)
void stats2_kernel(const float* __restrict__ y0, const float* __restrict__ y1,
                   const int* __restrict__ batch, float* __restrict__ stats) {
  const int n = blockIdx.x, w = threadIdx.x;
  const int g = batch[n];
  const float mean0 = stats[kG + g];
  float f = y0[(size_t)n * 128 + w] - mean0;
  float s0 = f * f;
  float s1 = 0.f;
#pragma unroll
  for (int i = 0; i < 3; ++i) {
    float t = y1[((size_t)i * kN + n) * 128 + w];
    s1 = fmaf(t, t, s1);
  }
  __shared__ float r0[128], r1[128];
  r0[w] = s0; r1[w] = s1;
  __syncthreads();
  for (int s = 64; s > 0; s >>= 1) {
    if (w < s) { r0[w] += r0[w + s]; r1[w] += r1[w + s]; }
    __syncthreads();
  }
  if (w == 0) {
    atomicAdd(&stats[2 * kG + g], r0[0]);
    atomicAdd(&stats[3 * kG + g], r1[0]);
  }
}

__global__ void stats_fin_kernel(float* __restrict__ stats) {
  const int g = threadIdx.x;
  if (g >= kG) return;
  const float c = stats[g];
  const float var0 = stats[2 * kG + g] / (c * 128.f);
  const float var1 = stats[3 * kG + g] / (c * 384.f);
  stats[2 * kG + g] = 1.f / (sqrtf(var0) + kEPS);
  stats[3 * kG + g] = 1.f / (sqrtf(var1) + kEPS);
}

// ---------------------------------------------------------------------------
// final: normalization + residual skip GEMVs, interleaved output store.
// ---------------------------------------------------------------------------
__global__ __launch_bounds__(128)
void final_kernel(const float* __restrict__ nfh, const int* __restrict__ batch,
                  const float* __restrict__ y0, const float* __restrict__ y1,
                  const float* __restrict__ stats,
                  const float* __restrict__ ln_w0, const float* __restrict__ ln_b0,
                  const float* __restrict__ ln_w1,
                  const float* __restrict__ W_s0, const float* __restrict__ b_s0,
                  const float* __restrict__ W_s1, float* __restrict__ out) {
  const int n = blockIdx.x, w = threadIdx.x;
  __shared__ float h0_l[128], h1_l[3][128];
  h0_l[w] = nfh[(size_t)n * 512 + w];
#pragma unroll
  for (int i = 0; i < 3; ++i)
    h1_l[i][w] = nfh[(size_t)n * 512 + 128 + w * 3 + i];
  __syncthreads();

  const int g = batch[n];
  const float mean0 = stats[kG + g];
  const float rstd0 = stats[2 * kG + g];
  const float rstd1 = stats[3 * kG + g];

  float acc0 = 0.f, acc1[3] = {0.f, 0.f, 0.f};
  for (int u = 0; u < 128; ++u) {
    const float ws0 = W_s0[(size_t)u * 128 + w];
    const float ws1 = W_s1[(size_t)u * 128 + w];
    acc0 = fmaf(h0_l[u], ws0, acc0);
#pragma unroll
    for (int i = 0; i < 3; ++i) acc1[i] = fmaf(h1_l[i][u], ws1, acc1[i]);
  }
  const float f0 = (y0[(size_t)n * 128 + w] - mean0) * rstd0 * ln_w0[w] + ln_b0[w];
  out[(size_t)n * 512 + w] = f0 + acc0 * INV_H + b_s0[w];
#pragma unroll
  for (int i = 0; i < 3; ++i) {
    const float f1 = y1[((size_t)i * kN + n) * 128 + w] * rstd1 * ln_w1[w];
    out[(size_t)n * 512 + 128 + w * 3 + i] = f1 + acc1[i] * INV_H;
  }
}

// ---------------------------------------------------------------------------
extern "C" void kernel_launch(void* const* d_in, const int* in_sizes, int n_in,
                              void* d_out, int out_size, void* d_ws, size_t ws_size,
                              hipStream_t stream) {
  (void)in_sizes; (void)n_in; (void)out_size; (void)ws_size;

  const float* nfh    = (const float*)d_in[0];
  const float* nfu    = (const float*)d_in[1];
  const float* onehot = (const float*)d_in[2];
  const int*   batch  = (const int*)d_in[3];
  const float* W_h0   = (const float*)d_in[4];
  const float* W_h1   = (const float*)d_in[5];
  const float* W_u0   = (const float*)d_in[6];
  const float* W_u1   = (const float*)d_in[7];
  const float* Wt_00  = (const float*)d_in[8];
  const float* Wt_11  = (const float*)d_in[9];
  const float* Wt_01  = (const float*)d_in[10];
  const float* Wt_10  = (const float*)d_in[11];
  const float* W_p0   = (const float*)d_in[12];
  const float* b_p0   = (const float*)d_in[13];
  const float* W_p1   = (const float*)d_in[14];
  const float* Wsc0   = (const float*)d_in[15];
  const float* Wsc1   = (const float*)d_in[16];
  const float* ln_w0  = (const float*)d_in[17];
  const float* ln_b0  = (const float*)d_in[18];
  const float* ln_w1  = (const float*)d_in[19];
  const float* W_s0   = (const float*)d_in[20];
  const float* b_s0   = (const float*)d_in[21];
  const float* W_s1   = (const float*)d_in[22];
  float* out = (float*)d_out;

  // workspace carve-up (f32 region, then swizzled f16 weights region)
  float* ws   = (float*)d_ws;
  float* hp0  = ws;                         // N*128
  float* up0  = hp0 + (size_t)kN * 128;     // N*64
  float* hp1  = up0 + (size_t)kN * 64;      // 3*N*128  ([i][n][u])
  float* up1  = hp1 + (size_t)3 * kN * 128; // 3*N*64
  float* z0   = up1 + (size_t)3 * kN * 64;  // N*256
  float* z1   = z0  + (size_t)kN * 256;     // 3*N*128
  float* y0   = z1  + (size_t)3 * kN * 128; // N*128
  float* y1   = y0  + (size_t)kN * 128;     // 3*N*128
  float* stats = y1 + (size_t)3 * kN * 128; // 4*G
  _Float16* W00s = (_Float16*)(stats + 4 * kG);
  _Float16* W11s = W00s + (size_t)128 * 64 * 256;
  _Float16* W01s = W11s + (size_t)128 * 64 * 256;
  _Float16* W10s = W01s + (size_t)128 * 64 * 128;

  // 1) weights -> f16, swizzled into per-fragment-contiguous layout
  const int nW00 = 128 * 64 * 256, nW01 = 128 * 64 * 128;
  cvt_swz_kernel<<<(nW00 + 255) / 256, 256, 0, stream>>>(Wt_00, W00s, 256, nW00);
  cvt_swz_kernel<<<(nW00 + 255) / 256, 256, 0, stream>>>(Wt_11, W11s, 256, nW00);
  cvt_swz_kernel<<<(nW01 + 255) / 256, 256, 0, stream>>>(Wt_01, W01s, 128, nW01);
  cvt_swz_kernel<<<(nW01 + 255) / 256, 256, 0, stream>>>(Wt_10, W10s, 128, nW01);

  // 2) projections (f32 WMMA, full precision)
  dim3 gh(kN / 16, 128 / 16);
  proj_kernel<<<gh, 32, 0, stream>>>(nfh, 512, 1, W_h0, hp0, 128, 128, INV_H);
  for (int i = 0; i < 3; ++i)
    proj_kernel<<<gh, 32, 0, stream>>>(nfh + 128 + i, 512, 3, W_h1,
                                       hp1 + (size_t)i * kN * 128, 128, 128, INV_H);
  dim3 gu(kN / 16, 64 / 16);
  proj_kernel<<<gu, 32, 0, stream>>>(nfu, 256, 1, W_u0, up0, 64, 64, INV_U);
  for (int i = 0; i < 3; ++i)
    proj_kernel<<<gu, 32, 0, stream>>>(nfu + 64 + i, 256, 3, W_u1,
                                       up1 + (size_t)i * kN * 64, 64, 64, INV_U);

  // 3) heavy tensor-product GEMMs (f16 WMMA, f32 accumulate)
  z0_kernel<<<kN / 16, 256, 0, stream>>>(hp0, up0, hp1, up1, W00s, W11s, z0);
  dim3 gz1(kN / 16, 3);
  z1_kernel<<<gz1, 256, 0, stream>>>(hp0, up0, hp1, up1, W01s, W10s, z1);

  // 4) gating + projections + species scatter
  post_kernel<<<kN, 128, 0, stream>>>(nfh, onehot, z0, z1, W_p0, b_p0, W_p1,
                                      Wsc0, Wsc1, y0, y1);

  // 5) group statistics
  stats_zero_kernel<<<1, 256, 0, stream>>>(stats);
  stats1_kernel<<<kN, 128, 0, stream>>>(y0, batch, stats);
  stats_mid_kernel<<<1, 64, 0, stream>>>(stats);
  stats2_kernel<<<kN, 128, 0, stream>>>(y0, y1, batch, stats);
  stats_fin_kernel<<<1, 64, 0, stream>>>(stats);

  // 6) normalize + residual skip, write interleaved output
  final_kernel<<<kN, 128, 0, stream>>>(nfh, batch, y0, y1, stats,
                                       ln_w0, ln_b0, ln_w1, W_s0, b_s0, W_s1, out);
}
